// ShiftedWindowMSA_83210696393647
// MI455X (gfx1250) — matrix operand order repaired and checked by reading, not compiled, not run on hardware
//
#include <hip/hip_runtime.h>
#include <hip/hip_bf16.h>
#include <math.h>

typedef _Float16     v16h  __attribute__((ext_vector_type(16)));
typedef _Float16     v8h   __attribute__((ext_vector_type(8)));
typedef float        v8f   __attribute__((ext_vector_type(8)));
typedef float        f32x4 __attribute__((ext_vector_type(4)));
typedef unsigned int u32x4 __attribute__((ext_vector_type(4)));

#define NIMG  32
#define PIX   3136
#define CDIM  128
#define NWT   49

__device__ __forceinline__ v8f wmma16(v16h a, v16h b, v8f c) {
  // D = A(16x32 f16) * B(32x16 f16) + C(16x16 f32)
  return __builtin_amdgcn_wmma_f32_16x16x32_f16(false, a, false, b, (short)0, c,
                                                false, false);
}

__device__ __forceinline__ v16h cat16(v8h lo, v8h hi) {
  return __builtin_shufflevector(lo, hi, 0,1,2,3,4,5,6,7,8,9,10,11,12,13,14,15);
}

// Build f16 A-fragment from fp32 LDS row: K = {h*8+e} and {16+h*8+e}.
__device__ __forceinline__ v16h afrag_from_f32(const float* rp) {
  f32x4 f0 = *(const f32x4*)&rp[0];
  f32x4 f1 = *(const f32x4*)&rp[4];
  f32x4 f2 = *(const f32x4*)&rp[16];
  f32x4 f3 = *(const f32x4*)&rp[20];
  v16h a;
#pragma unroll
  for (int i = 0; i < 4; ++i) {
    a[i]      = (_Float16)f0[i];
    a[4 + i]  = (_Float16)f1[i];
    a[8 + i]  = (_Float16)f2[i];
    a[12 + i] = (_Float16)f3[i];
  }
  return a;
}

// ---------------------------------------------------------------------------
// Kernel 1: pack fp32 weights into f16 WMMA B-fragment layout.
// B-frag layout (16x16x32 f16): lane = 16h+n holds B[k = kt*32 + h*8 + (e&7) + 16*(e>>3)][n],
// e = 0..15.  B[k][n] = W[col=n][k]  (since y = x @ W^T).
// ---------------------------------------------------------------------------
__global__ __launch_bounds__(256) void pack_weights(
    const float* __restrict__ qkv_w, const float* __restrict__ out_w,
    _Float16* __restrict__ wq, _Float16* __restrict__ wo)
{
  int i = blockIdx.x * 256 + threadIdx.x;
  if (i < 4 * 24 * 32 * 16) {            // qkv_w: K=128 (4 kt) x N=384 (24 nt)
    int e    = i & 15;
    int lane = (i >> 4) & 31;
    int tile = i >> 9;                   // kt*24 + nt
    int nt = tile % 24, kt = tile / 24;
    int n = lane & 15, hh = lane >> 4;
    wq[i] = (_Float16)qkv_w[(size_t)(nt * 16 + n) * 128 + kt * 32 + hh * 8 + (e & 7) + 16 * (e >> 3)];
  }
  if (i < 4 * 8 * 32 * 16) {             // out_w: K=128 (4 kt) x N=128 (8 nt)
    int e    = i & 15;
    int lane = (i >> 4) & 31;
    int tile = i >> 9;                   // kt*8 + nt
    int nt = tile & 7, kt = tile >> 3;
    int n = lane & 15, hh = lane >> 4;
    wo[i] = (_Float16)out_w[(size_t)(nt * 16 + n) * 128 + kt * 32 + hh * 8 + (e & 7) + 16 * (e >> 3)];
  }
}

// ---------------------------------------------------------------------------
// Kernel 2: QKV projection.  [100352 x 128] fp32 @ packed-W -> f16 [m][384]
// Block: 64 rows, 256 threads (8 waves).  Wave w: mt = w&3, nt in 12-tile strip.
// ---------------------------------------------------------------------------
__global__ __launch_bounds__(256) void qkv_gemm(
    const float* __restrict__ x, const float* __restrict__ qkv_b,
    const _Float16* __restrict__ wpack, _Float16* __restrict__ qkv)
{
  __shared__ float xs[64][132];          // 132: 16B-aligned rows, bank stride 4
  const int tid = threadIdx.x;
  const size_t block_m = (size_t)blockIdx.x * 64;

  { // stage 64x128 fp32 tile (x is read-once: non-temporal)
    const int row = tid >> 2, part = tid & 3;
    const f32x4* src = (const f32x4*)(x + (block_m + row) * 128 + part * 32);
    f32x4* dst = (f32x4*)&xs[row][part * 32];
#pragma unroll
    for (int v = 0; v < 8; ++v) dst[v] = __builtin_nontemporal_load(src + v);
  }
  __syncthreads();

  const int wave = tid >> 5, lane = tid & 31;
  const int h = lane >> 4, ln = lane & 15;
  const int mt = wave & 3;
  const int nt0 = (wave >> 2) * 12;

  v8f acc[12] = {};
  for (int kt = 0; kt < 4; ++kt) {
    v16h a = afrag_from_f32(&xs[mt * 16 + ln][kt * 32 + h * 8]);
    // prefetch all 12 B fragments: one base, 1024B constant strides
    const _Float16* wbase = wpack + ((size_t)(kt * 24 + nt0) * 32 + lane) * 16;
    v16h b[12];
#pragma unroll
    for (int j = 0; j < 12; ++j) b[j] = *(const v16h*)(wbase + (size_t)j * 512);
#pragma unroll
    for (int j = 0; j < 12; ++j) acc[j] = wmma16(a, b[j], acc[j]);
  }

  float bb[12];
#pragma unroll
  for (int j = 0; j < 12; ++j) bb[j] = qkv_b[(nt0 + j) * 16 + ln];
#pragma unroll
  for (int r = 0; r < 8; ++r) {          // D: VGPR r -> row r + 8h
    size_t m = block_m + mt * 16 + r + 8 * h;
    _Float16* qrow = qkv + m * 384 + nt0 * 16 + ln;
#pragma unroll
    for (int j = 0; j < 12; ++j)         // adjacent 32B runs -> L2 line merge
      qrow[j * 16] = (_Float16)(acc[j][r] + bb[j]);
  }
}

// ---------------------------------------------------------------------------
// Kernel 3: windowed attention.  One block per (img, head, wy, wx); 128 thr.
// Reproduces reference reshape semantics + shift roll exactly.
// ---------------------------------------------------------------------------
__global__ __launch_bounds__(128) void attn_kernel(
    const _Float16* __restrict__ qkv, const float* __restrict__ rpe,
    _Float16* __restrict__ ao)
{
  __shared__ _Float16 Qs[64][40];
  __shared__ _Float16 Ks[64][40];
  __shared__ _Float16 Vts[32][72];       // V transposed: [d][token]
  __shared__ float    Ss[64][65];
  __shared__ _Float16 Ps[64][72];
  __shared__ float    rpes[169];

  const int tid = threadIdx.x;
  const unsigned bid = blockIdx.x;
  const int img  = bid >> 8;
  const int head = (bid >> 6) & 3;
  const int wy   = (bid >> 3) & 7;
  const int wx   = bid & 7;

  for (int i = tid; i < 169; i += 128) rpes[i] = rpe[i];

  { // gather window: shift-roll + reference reshape mapping, de-interleave qkv
    const int t = tid >> 1;              // token 0..63
    const int part = tid & 1;            // d range [16*part, 16*part+16)
    if (t < NWT) {
      unsigned flat = head * 100352u + wy * 12544u + wx * 1568u + (unsigned)t * 32u;
      unsigned y  = flat / 7168u;
      unsigned xx = (flat / 128u) % 56u;
      unsigned c0 = flat % 128u;         // multiple of 32; d-range stays in-block
      unsigned ys = (y + 3u) % 56u;      // roll(-3,-3): window (y,x) <- orig (y+3,x+3)
      unsigned xsrc = (xx + 3u) % 56u;
      size_t base = ((size_t)img * PIX + ys * 56u + xsrc) * 384u + 3u * (c0 + 16u * part);
      union { u32x4 v[6]; _Float16 hv[48]; } buf;
      const u32x4* src = (const u32x4*)(qkv + base);
#pragma unroll
      for (int i = 0; i < 6; ++i) buf.v[i] = __builtin_nontemporal_load(src + i);
#pragma unroll
      for (int i = 0; i < 16; ++i) {
        int d = 16 * part + i;
        Qs[t][d]  = buf.hv[3 * i + 0];
        Ks[t][d]  = buf.hv[3 * i + 1];
        Vts[d][t] = buf.hv[3 * i + 2];
      }
    } else {
#pragma unroll
      for (int i = 0; i < 16; ++i) {
        int d = 16 * part + i;
        Qs[t][d] = (_Float16)0.f;
        Ks[t][d] = (_Float16)0.f;
        Vts[d][t] = (_Float16)0.f;
      }
    }
  }
  __syncthreads();

  const int wave = tid >> 5, lane = tid & 31;
  const int h = lane >> 4, ln = lane & 15;
  const int mt = wave;                   // each wave owns one 16-row M tile

  { // S = Q K^T : 4 back-to-back WMMAs, then one fused epilogue
    v8h alo = *(const v8h*)&Qs[mt * 16 + ln][h * 8];
    v8h ahi = *(const v8h*)&Qs[mt * 16 + ln][16 + h * 8];
    v16h a = cat16(alo, ahi);
    v16h bfr[4];
#pragma unroll
    for (int nt = 0; nt < 4; ++nt) {
      v8h blo = *(const v8h*)&Ks[nt * 16 + ln][h * 8];
      v8h bhi = *(const v8h*)&Ks[nt * 16 + ln][16 + h * 8];
      bfr[nt] = cat16(blo, bhi);
    }
    v8f acc[4] = {};
#pragma unroll
    for (int nt = 0; nt < 4; ++nt) acc[nt] = wmma16(a, bfr[nt], acc[nt]);

#pragma unroll
    for (int nt = 0; nt < 4; ++nt) {
      const int tk = nt * 16 + ln;
#pragma unroll
      for (int r = 0; r < 8; ++r) {
        int tq = mt * 16 + r + 8 * h;
        float v;
        if (tq < NWT && tk < NWT) {
          v = acc[nt][r] * 0.17677669529663687f;  // 1/sqrt(32)
          int qy = tq / 7, qx = tq % 7;
          int ky = tk / 7, kx = tk % 7;
          v += rpes[(ky - qy + 6) * 13 + (kx - qx + 6)];
          if (wy == 7 && ((tq >= 28) != (tk >= 28))) v = -INFINITY;
          if (wx == 7 && (((qx == 4) && (kx < 4)) || ((qx < 4) && (kx >= 4)))) v = -INFINITY;
        } else {
          v = -INFINITY;                 // pad rows/cols
        }
        Ss[tq][tk] = v;
      }
    }
  }
  __syncthreads();

  if (tid < 64) {                        // row softmax (64-wide, padded)
    const int row = tid;
    float mx = -INFINITY;
    for (int k = 0; k < 64; ++k) mx = fmaxf(mx, Ss[row][k]);
    float sum = 0.f;
    for (int k = 0; k < 64; ++k) {
      float s = Ss[row][k];
      float e = (s == -INFINITY) ? 0.f : __expf(s - mx);
      Ss[row][k] = e;
      sum += e;
    }
    float rinv = (sum > 0.f) ? (1.f / sum) : 0.f;
    for (int k = 0; k < 64; ++k) Ps[row][k] = (_Float16)(Ss[row][k] * rinv);
  }
  __syncthreads();

  // O = P V  (K dim = 64 tokens -> 2 k-tiles; N = 32 head dims -> 2 n-tiles)
  v16h pa[2];
  v16h vb[2][2];
#pragma unroll
  for (int kt = 0; kt < 2; ++kt) {
    pa[kt] = cat16(*(const v8h*)&Ps[mt * 16 + ln][kt * 32 + h * 8],
                   *(const v8h*)&Ps[mt * 16 + ln][kt * 32 + 16 + h * 8]);
#pragma unroll
    for (int nd = 0; nd < 2; ++nd)
      vb[kt][nd] = cat16(*(const v8h*)&Vts[nd * 16 + ln][kt * 32 + h * 8],
                         *(const v8h*)&Vts[nd * 16 + ln][kt * 32 + 16 + h * 8]);
  }
  v8f oacc[2] = {};
#pragma unroll
  for (int kt = 0; kt < 2; ++kt)
#pragma unroll
    for (int nd = 0; nd < 2; ++nd) oacc[nd] = wmma16(pa[kt], vb[kt][nd], oacc[nd]);

  // un-window + roll(+3,+3) scatter to planar f16 [img][pix][c]
#pragma unroll
  for (int r = 0; r < 8; ++r) {
    int t = mt * 16 + r + 8 * h;
    if (t < NWT) {
      unsigned flat = head * 100352u + wy * 12544u + wx * 1568u + (unsigned)t * 32u + (unsigned)ln;
      unsigned y  = flat / 7168u;
      unsigned xx = (flat / 128u) % 56u;
      unsigned c  = flat % 128u;         // d = ln; nd adds +16 within same block
      unsigned yf = (y + 3u) % 56u;
      unsigned xf = (xx + 3u) % 56u;
      _Float16* dst = ao + ((size_t)img * PIX + yf * 56u + xf) * 128u + c;
      dst[0]  = (_Float16)oacc[0][r];
      dst[16] = (_Float16)oacc[1][r];
    }
  }
}

// ---------------------------------------------------------------------------
// Kernel 4: output projection.  f16 [100352 x 128] @ packed out_w -> fp32.
// ---------------------------------------------------------------------------
__global__ __launch_bounds__(256) void proj_gemm(
    const _Float16* __restrict__ ao, const float* __restrict__ out_b,
    const _Float16* __restrict__ wpack, float* __restrict__ out)
{
  __shared__ _Float16 xs[64][136];       // 272B rows: 16B aligned, bank stride 4
  const int tid = threadIdx.x;
  const size_t block_m = (size_t)blockIdx.x * 64;
  { // ao is read-once: non-temporal
    const int row = tid >> 2, part = tid & 3;
    const u32x4* src = (const u32x4*)(ao + (block_m + row) * 128 + part * 32);
    u32x4* dst = (u32x4*)&xs[row][part * 32];
#pragma unroll
    for (int v = 0; v < 4; ++v) dst[v] = __builtin_nontemporal_load(src + v);
  }
  __syncthreads();

  const int wave = tid >> 5, lane = tid & 31;
  const int h = lane >> 4, ln = lane & 15;
  const int mt = wave & 3;
  const int nt0 = (wave >> 2) * 4;

  v8f acc[4] = {};
  for (int kt = 0; kt < 4; ++kt) {
    v16h a = cat16(*(const v8h*)&xs[mt * 16 + ln][kt * 32 + h * 8],
                   *(const v8h*)&xs[mt * 16 + ln][kt * 32 + 16 + h * 8]);
    const _Float16* wbase = wpack + ((size_t)(kt * 8 + nt0) * 32 + lane) * 16;
    v16h b[4];
#pragma unroll
    for (int j = 0; j < 4; ++j) b[j] = *(const v16h*)(wbase + (size_t)j * 512);
#pragma unroll
    for (int j = 0; j < 4; ++j) acc[j] = wmma16(a, b[j], acc[j]);
  }

  float bb[4];
#pragma unroll
  for (int j = 0; j < 4; ++j) bb[j] = out_b[(nt0 + j) * 16 + ln];
#pragma unroll
  for (int r = 0; r < 8; ++r) {
    size_t m = block_m + mt * 16 + r + 8 * h;
    float* orow = out + m * 128 + nt0 * 16 + ln;
#pragma unroll
    for (int j = 0; j < 4; ++j)          // final output: write-once -> NT store
      __builtin_nontemporal_store(acc[j][r] + bb[j], orow + j * 16);
  }
}

// ---------------------------------------------------------------------------
extern "C" void kernel_launch(void* const* d_in, const int* in_sizes, int n_in,
                              void* d_out, int out_size, void* d_ws, size_t ws_size,
                              hipStream_t stream)
{
  (void)in_sizes; (void)n_in; (void)out_size; (void)ws_size;
  const float* x     = (const float*)d_in[0];
  const float* qkv_w = (const float*)d_in[1];
  const float* qkv_b = (const float*)d_in[2];
  const float* out_w = (const float*)d_in[3];
  const float* out_b = (const float*)d_in[4];
  const float* rpe   = (const float*)d_in[5];
  float* out = (float*)d_out;

  char* ws = (char*)d_ws;
  // 32*3136*384 f16 = 77,070,336 B ; 32*3136*128 f16 = 25,690,112 B
  _Float16* qkv = (_Float16*)(ws);
  _Float16* ao  = (_Float16*)(ws + 77070336);
  _Float16* wq  = (_Float16*)(ws + 77070336 + 25690112);
  _Float16* wo  = (_Float16*)(ws + 77070336 + 25690112 + 98304);

  pack_weights<<<192,  256, 0, stream>>>(qkv_w, out_w, wq, wo);
  qkv_gemm    <<<1568, 256, 0, stream>>>(x, qkv_b, wq, qkv);
  attn_kernel <<<8192, 128, 0, stream>>>(qkv, rpe, ao);
  proj_gemm   <<<1568, 256, 0, stream>>>(ao, out_b, wo, out);
}
